// DecoderRNN_40604620817021
// MI455X (gfx1250) — matrix operand order repaired and checked by reading, not compiled, not run on hardware
//
#include <hip/hip_runtime.h>
#include <hip/hip_bf16.h>

// ---------------------------------------------------------------------------
// DecoderRNN for MI455X (gfx1250): bf16 WMMA GEMMs + per-step LSTM scan.
// ---------------------------------------------------------------------------

typedef __bf16 bf16_t;
typedef __attribute__((ext_vector_type(16))) __bf16 v16bf;
typedef __attribute__((ext_vector_type(8)))  __bf16 v8bf;
typedef __attribute__((ext_vector_type(8)))  float  v8f;

#define BQ   16
#define TQ   128
#define HQ   512
#define G4H  2048
#define VQ   32000

__device__ __forceinline__ float sigmoidf_(float x) {
  return 1.0f / (1.0f + __expf(-x));
}

// A fragment: 16x32 bf16, row-major A[lda], tile at (m0, k0).
// ISA layout: lanes 0-15 -> M=lane, lanes 16-31 -> same M, upper K half.
// lane's elements: e<8 -> k = half*8 + e ; e>=8 -> k = 16 + half*8 + (e-8).
__device__ __forceinline__ v16bf load_a16x32(const bf16_t* __restrict__ A,
                                             int lda, int m0, int k0) {
  int lane = threadIdx.x & 31;
  int m    = m0 + (lane & 15);
  int half = lane >> 4;
  const bf16_t* p = A + (size_t)m * lda + k0 + half * 8;
  v8bf lo = *(const v8bf*)p;
  v8bf hi = *(const v8bf*)(p + 16);
  v16bf a;
#pragma unroll
  for (int e = 0; e < 8; ++e) { a[e] = lo[e]; a[8 + e] = hi[e]; }
  return a;
}

// B fragment: 32x16 bf16 where B(k,n) = W[n][k], W row-major [N, ldw].
// ISA layout: lane half h holds e<8 -> k = h*8 + e ; e>=8 -> k = 16 + h*8 + (e-8).
__device__ __forceinline__ v16bf load_b32x16(const bf16_t* __restrict__ W,
                                             int ldw, int n0, int k0) {
  int lane = threadIdx.x & 31;
  int n    = n0 + (lane & 15);
  int kb   = k0 + (lane >> 4) * 8;
  const bf16_t* p = W + (size_t)n * ldw + kb;
  v8bf lo = *(const v8bf*)p;
  v8bf hi = *(const v8bf*)(p + 16);
  v16bf b;
#pragma unroll
  for (int e = 0; e < 8; ++e) { b[e] = lo[e]; b[8 + e] = hi[e]; }
  return b;
}

// ---------------------------------------------------------------------------
// f32 -> bf16 weight cast (weights stay L2-resident afterwards)
__global__ void f32_to_bf16_kernel(const float* __restrict__ src,
                                   bf16_t* __restrict__ dst, size_t n) {
  size_t i = (size_t)blockIdx.x * blockDim.x + threadIdx.x;
  size_t stride = (size_t)gridDim.x * blockDim.x;
  for (; i < n; i += stride) dst[i] = (bf16_t)src[i];
}

// Embedding gather + ReLU -> x_bf [T][B][H] bf16. SOS = 0 shifted teacher forcing.
__global__ void embed_kernel(const int* __restrict__ tgt,
                             const float* __restrict__ emb,
                             bf16_t* __restrict__ xbf) {
  size_t i = (size_t)blockIdx.x * 256 + threadIdx.x;  // over T*B*H = 1048576
  int k = (int)(i & (HQ - 1));
  int b = (int)((i >> 9) & 15);
  int t = (int)(i >> 13);
  int tok = (t == 0) ? 0 : tgt[b * TQ + (t - 1)];
  float v = emb[(size_t)tok * HQ + k];
  xbf[i] = (bf16_t)fmaxf(v, 0.0f);
}

// Initial states: h0 -> bf16 double-buffer slot 0, c0 -> f32 state.
__global__ void init_state_kernel(const float* __restrict__ h0,
                                  const float* __restrict__ c0,
                                  bf16_t* __restrict__ h_l0, bf16_t* __restrict__ h_l1,
                                  float* __restrict__ c_l0, float* __restrict__ c_l1) {
  int i = blockIdx.x * 256 + threadIdx.x;   // 2*16*512 = 16384
  int layer = i >> 13;
  int idx   = i & 8191;
  if (layer == 0) { h_l0[idx] = (bf16_t)h0[i]; c_l0[idx] = c0[i]; }
  else            { h_l1[idx] = (bf16_t)h0[i]; c_l1[idx] = c0[i]; }
}

// ---------------------------------------------------------------------------
// xp = A[2048,512](bf16) @ W[2048,512]^T (bf16) + b1 + b2  -> f32 [2048,2048]
// One 16x16 tile per wave, 16 K-steps of v_wmma_f32_16x16x32_bf16.
__global__ void gemm_xp_kernel(const bf16_t* __restrict__ A,
                               const bf16_t* __restrict__ W,
                               const float* __restrict__ b1,
                               const float* __restrict__ b2,
                               float* __restrict__ C) {
  int wid  = threadIdx.x >> 5;
  int lane = threadIdx.x & 31;
  int tile = blockIdx.x * 4 + wid;
  const int ntile = G4H / 16;       // 128
  int mt = tile / ntile, nt = tile - mt * ntile;
  int m0 = mt * 16, n0 = nt * 16;
  v8f acc = {0.f, 0.f, 0.f, 0.f, 0.f, 0.f, 0.f, 0.f};
#pragma unroll 4
  for (int k0 = 0; k0 < HQ; k0 += 32) {
    v16bf a = load_a16x32(A, HQ, m0, k0);
    v16bf b = load_b32x16(W, HQ, n0, k0);
    acc = __builtin_amdgcn_wmma_f32_16x16x32_bf16(false, a, false, b,
                                                  (short)0, acc, false, false);
  }
  int half = lane >> 4;
  int n    = n0 + (lane & 15);
  float bias = b1[n] + b2[n];
#pragma unroll
  for (int r = 0; r < 8; ++r) {
    int m = m0 + r + 8 * half;
    C[(size_t)m * G4H + n] = acc[r] + bias;
  }
}

// ---------------------------------------------------------------------------
// One LSTM timestep. 32 waves (8 blocks x 4 waves); wave owns 16 h-columns
// and computes its i/f/g/o tiles: gates = xp[t] + h_prev @ w_hh^T.
// h state double-buffered (cross-workgroup RAW); c state is slice-private.
__global__ void lstm_step_kernel(const float* __restrict__ xp,      // [T*B, 4H]
                                 const bf16_t* __restrict__ whh,    // [4H, H] bf16
                                 const bf16_t* __restrict__ h_src,  // [B, H]
                                 bf16_t* __restrict__ h_dst,        // [B, H]
                                 float* __restrict__ c_st,          // [B, H]
                                 bf16_t* __restrict__ hseq,         // [T, B, H] bf16
                                 float* __restrict__ hT,            // [B, H] f32 (d_out)
                                 float* __restrict__ cT,            // [B, H] f32 (d_out)
                                 int t) {
  int wid  = threadIdx.x >> 5;
  int lane = threadIdx.x & 31;
  int slice = blockIdx.x * 4 + wid;   // 0..31
  int j0   = slice * 16;
  int half = lane >> 4;
  int nl   = lane & 15;

  v8f acc[4];
  const float* xprow = xp + (size_t)(t * BQ) * G4H;
#pragma unroll
  for (int g = 0; g < 4; ++g)
#pragma unroll
    for (int r = 0; r < 8; ++r)
      acc[g][r] = xprow[(size_t)(r + 8 * half) * G4H + g * HQ + j0 + nl];

  for (int k0 = 0; k0 < HQ; k0 += 32) {
    v16bf a = load_a16x32(h_src, HQ, 0, k0);   // shared across the 4 gates
#pragma unroll
    for (int g = 0; g < 4; ++g) {
      v16bf b = load_b32x16(whh, HQ, g * HQ + j0, k0);
      acc[g] = __builtin_amdgcn_wmma_f32_16x16x32_bf16(false, a, false, b,
                                                       (short)0, acc[g], false, false);
    }
  }

#pragma unroll
  for (int r = 0; r < 8; ++r) {
    int m   = r + 8 * half;          // batch index
    int col = j0 + nl;
    float iv = sigmoidf_(acc[0][r]);
    float fv = sigmoidf_(acc[1][r]);
    float gv = tanhf(acc[2][r]);
    float ov = sigmoidf_(acc[3][r]);
    size_t idx = (size_t)m * HQ + col;
    float cn = fv * c_st[idx] + iv * gv;
    float hn = ov * tanhf(cn);
    c_st[idx] = cn;
    h_dst[idx] = (bf16_t)hn;
    hseq[(size_t)t * BQ * HQ + idx] = (bf16_t)hn;
    if (t == TQ - 1) { hT[idx] = hn; cT[idx] = cn; }
  }
}

// ---------------------------------------------------------------------------
// logits[b][t][v] = h2s[t*B+b,:] @ w_out[v,:] + b_out[v].
// 4x4 register-blocked: each wave computes a 64x64 output block (16 WMMA
// accumulators), cutting per-tile A and B re-reads 4x each (~8.2 GB -> ~2 GB
// of L2 traffic). Output staged through LDS -> 128B contiguous per-lane
// stores into the [B,T,V] layout (m-tile == one timestep, rows == batch).
__global__ void gemm_logits_kernel(const bf16_t* __restrict__ A,   // [2048, 512]
                                   const bf16_t* __restrict__ W,   // [32000, 512]
                                   const float* __restrict__ bias, // [32000]
                                   float* __restrict__ out) {      // [B, T, V]
  __shared__ __align__(16) float stage[4][16][68];
  int wid  = threadIdx.x >> 5;
  int lane = threadIdx.x & 31;
  const int ngrp = VQ / 64;         // 500 column groups of 64
  int tile = blockIdx.x * 4 + wid;  // 0..15999
  int mg = tile / ngrp;             // 0..31  (64 rows = 4 timesteps)
  int ng = tile - mg * ngrp;        // 0..499 (64 vocab columns)
  int m0 = mg * 64, n0 = ng * 64;

  v8f acc[4][4];
#pragma unroll
  for (int i = 0; i < 4; ++i)
#pragma unroll
    for (int j = 0; j < 4; ++j)
      acc[i][j] = (v8f){0.f, 0.f, 0.f, 0.f, 0.f, 0.f, 0.f, 0.f};

  for (int k0 = 0; k0 < HQ; k0 += 32) {
    v16bf bf[4];
#pragma unroll
    for (int j = 0; j < 4; ++j) bf[j] = load_b32x16(W, HQ, n0 + j * 16, k0);
#pragma unroll
    for (int i = 0; i < 4; ++i) {
      v16bf a = load_a16x32(A, HQ, m0 + i * 16, k0);
#pragma unroll
      for (int j = 0; j < 4; ++j)
        acc[i][j] = __builtin_amdgcn_wmma_f32_16x16x32_bf16(
            false, a, false, bf[j], (short)0, acc[i][j], false, false);
    }
  }

  int half = lane >> 4;
  int nl   = lane & 15;
  float bv[4];
#pragma unroll
  for (int j = 0; j < 4; ++j) bv[j] = bias[n0 + j * 16 + nl];

#pragma unroll 1
  for (int i = 0; i < 4; ++i) {
    int t = (m0 >> 4) + i;          // timestep of this m-row of tiles
#pragma unroll
    for (int j = 0; j < 4; ++j)
#pragma unroll
      for (int r = 0; r < 8; ++r)
        stage[wid][r + 8 * half][j * 16 + nl] = acc[i][j][r] + bv[j];
    __syncthreads();
    int row = lane >> 1;            // batch index b
    int seg = (lane & 1) * 32;
    float* dst = out + ((size_t)row * TQ + t) * VQ + n0 + seg;
#pragma unroll
    for (int e = 0; e < 32; e += 4)
      *(float4*)(dst + e) = *(const float4*)&stage[wid][row][seg + e];
    __syncthreads();
  }
}

// In-place log-softmax along V for each (b,t) row of d_out.
__global__ void log_softmax_kernel(float* __restrict__ out) {
  __shared__ float sm[8];
  __shared__ float sbc;
  int tid = threadIdx.x;
  float* row = out + (size_t)blockIdx.x * VQ;

  float mx = -3.0e38f;
  for (int i = tid; i < VQ; i += 256) mx = fmaxf(mx, row[i]);
#pragma unroll
  for (int o = 16; o > 0; o >>= 1) mx = fmaxf(mx, __shfl_down(mx, o, 32));
  if ((tid & 31) == 0) sm[tid >> 5] = mx;
  __syncthreads();
  if (tid == 0) {
    float m = sm[0];
#pragma unroll
    for (int i = 1; i < 8; ++i) m = fmaxf(m, sm[i]);
    sbc = m;
  }
  __syncthreads();
  mx = sbc;

  float s = 0.0f;
  for (int i = tid; i < VQ; i += 256) s += __expf(row[i] - mx);
#pragma unroll
  for (int o = 16; o > 0; o >>= 1) s += __shfl_down(s, o, 32);
  if ((tid & 31) == 0) sm[tid >> 5] = s;
  __syncthreads();
  if (tid == 0) {
    float tot = 0.0f;
#pragma unroll
    for (int i = 0; i < 8; ++i) tot += sm[i];
    sbc = mx + __logf(tot);
  }
  __syncthreads();
  float lse = sbc;
  for (int i = tid; i < VQ; i += 256) row[i] -= lse;
}

// ---------------------------------------------------------------------------
extern "C" void kernel_launch(void* const* d_in, const int* in_sizes, int n_in,
                              void* d_out, int out_size, void* d_ws, size_t ws_size,
                              hipStream_t stream) {
  (void)in_sizes; (void)n_in; (void)out_size; (void)ws_size;
  const float* h0    = (const float*)d_in[1];
  const float* c0    = (const float*)d_in[2];
  const int*   tgt   = (const int*)d_in[3];
  const float* emb   = (const float*)d_in[4];
  const float* w_ih0 = (const float*)d_in[5];
  const float* w_hh0 = (const float*)d_in[6];
  const float* b_ih0 = (const float*)d_in[7];
  const float* b_hh0 = (const float*)d_in[8];
  const float* w_ih1 = (const float*)d_in[9];
  const float* w_hh1 = (const float*)d_in[10];
  const float* b_ih1 = (const float*)d_in[11];
  const float* b_hh1 = (const float*)d_in[12];
  const float* w_out = (const float*)d_in[13];
  const float* b_out = (const float*)d_in[14];

  float* out  = (float*)d_out;
  float* logp = out;                                   // [B,T,V]
  float* hfin = out + (size_t)BQ * TQ * VQ;            // [2,B,H]
  float* cfin = hfin + 2 * BQ * HQ;                    // [2,B,H]

  char* base = (char*)d_ws;
  size_t off = 0;
  auto alloc = [&](size_t bytes) -> char* {
    off = (off + 255) & ~(size_t)255;
    char* p = base + off;
    off += bytes;
    return p;
  };

  const size_t WSZ = (size_t)G4H * HQ;          // 1,048,576 elems per LSTM weight
  bf16_t* wih0_b = (bf16_t*)alloc(WSZ * 2);
  bf16_t* whh0_b = (bf16_t*)alloc(WSZ * 2);
  bf16_t* wih1_b = (bf16_t*)alloc(WSZ * 2);
  bf16_t* whh1_b = (bf16_t*)alloc(WSZ * 2);
  bf16_t* wout_b = (bf16_t*)alloc((size_t)VQ * HQ * 2);
  bf16_t* x_bf   = (bf16_t*)alloc((size_t)TQ * BQ * HQ * 2);
  bf16_t* h1s_bf = (bf16_t*)alloc((size_t)TQ * BQ * HQ * 2);
  bf16_t* h2s_bf = (bf16_t*)alloc((size_t)TQ * BQ * HQ * 2);
  float*  xp     = (float*)alloc((size_t)TQ * BQ * G4H * 4);   // reused by both layers
  bf16_t* hbuf0  = (bf16_t*)alloc((size_t)2 * BQ * HQ * 2);    // layer0 dbl-buffer
  bf16_t* hbuf1  = (bf16_t*)alloc((size_t)2 * BQ * HQ * 2);    // layer1 dbl-buffer
  float*  cst0   = (float*)alloc((size_t)BQ * HQ * 4);
  float*  cst1   = (float*)alloc((size_t)BQ * HQ * 4);

  // Weight casts (L2-resident bf16 copies).
  f32_to_bf16_kernel<<<512, 256, 0, stream>>>(w_ih0, wih0_b, WSZ);
  f32_to_bf16_kernel<<<512, 256, 0, stream>>>(w_hh0, whh0_b, WSZ);
  f32_to_bf16_kernel<<<512, 256, 0, stream>>>(w_ih1, wih1_b, WSZ);
  f32_to_bf16_kernel<<<512, 256, 0, stream>>>(w_hh1, whh1_b, WSZ);
  f32_to_bf16_kernel<<<4096, 256, 0, stream>>>(w_out, wout_b, (size_t)VQ * HQ);

  init_state_kernel<<<64, 256, 0, stream>>>(h0, c0, hbuf0, hbuf1, cst0, cst1);
  embed_kernel<<<4096, 256, 0, stream>>>(tgt, emb, x_bf);

  // Layer 0: batched input-gate GEMM, then 128-step scan.
  gemm_xp_kernel<<<4096, 128, 0, stream>>>(x_bf, wih0_b, b_ih0, b_hh0, xp);
  for (int t = 0; t < TQ; ++t) {
    lstm_step_kernel<<<8, 128, 0, stream>>>(xp, whh0_b,
                                            hbuf0 + (t & 1) * (BQ * HQ),
                                            hbuf0 + ((t + 1) & 1) * (BQ * HQ),
                                            cst0, h1s_bf, hfin, cfin, t);
  }

  // Layer 1.
  gemm_xp_kernel<<<4096, 128, 0, stream>>>(h1s_bf, wih1_b, b_ih1, b_hh1, xp);
  for (int t = 0; t < TQ; ++t) {
    lstm_step_kernel<<<8, 128, 0, stream>>>(xp, whh1_b,
                                            hbuf1 + (t & 1) * (BQ * HQ),
                                            hbuf1 + ((t + 1) & 1) * (BQ * HQ),
                                            cst1, h2s_bf,
                                            hfin + BQ * HQ, cfin + BQ * HQ, t);
  }

  // Output projection + bias (4x4 blocked WMMA), then in-place log-softmax.
  gemm_logits_kernel<<<4000, 128, 0, stream>>>(h2s_bf, wout_b, b_out, logp);
  log_softmax_kernel<<<2048, 256, 0, stream>>>(logp);
}
